// DCNv2_89670327206155
// MI455X (gfx1250) — compile-verified, matrix operand order checked by reading, not run on hardware
//
#include <hip/hip_runtime.h>

typedef __attribute__((ext_vector_type(16))) _Float16 v16h;
typedef __attribute__((ext_vector_type(8)))  _Float16 v8h;
typedef __attribute__((ext_vector_type(8)))  float    v8f;

#define Bsz 8
#define Cch 128
#define Och 128
#define Hh 64
#define Ww 64
#define K2c 9
#define RED (Cch * K2c)   // 1152
#define NKC (RED / 32)    // 36 k-chunks of 32

// ---------------------------------------------------------------------------
// Kernel 1: 3x3 conv producing 27 channels (18 offsets + 9 mask logits),
// then softmax over the 9 mask channels. One thread per output position.
// Per-channel weight slice (27*9 floats) staged in LDS, shared by the block.
// ---------------------------------------------------------------------------
__global__ __launch_bounds__(256) void om_conv_kernel(
    const float* __restrict__ x, const float* __restrict__ w_om,
    const float* __restrict__ b_om,
    float* __restrict__ dyw, float* __restrict__ dxw, float* __restrict__ mkw)
{
    __shared__ float wsm[27 * 9];
    const int tid = threadIdx.x;
    const int p   = blockIdx.x * 256 + tid;
    const int b   = p >> 12;          // 4096 positions per image
    const int h   = (p >> 6) & 63;
    const int w   = p & 63;

    float acc[27];
    #pragma unroll
    for (int i = 0; i < 27; ++i) acc[i] = 0.f;

    const float* xb = x + (size_t)b * Cch * Hh * Ww;

    for (int c = 0; c < Cch; ++c) {
        __syncthreads();
        if (tid < 243) {
            int oc = tid / 9, k = tid % 9;
            // w_om layout: (27, 128, 3, 3)
            wsm[tid] = w_om[(size_t)oc * (Cch * 9) + c * 9 + k];
        }
        __syncthreads();
        const float* xc = xb + (size_t)c * Hh * Ww;
        float tap[9];
        #pragma unroll
        for (int k = 0; k < 9; ++k) {
            int yy = h - 1 + k / 3;
            int xx = w - 1 + k % 3;
            tap[k] = (yy >= 0 && yy < Hh && xx >= 0 && xx < Ww) ? xc[yy * Ww + xx] : 0.f;
        }
        #pragma unroll
        for (int oc = 0; oc < 27; ++oc) {
            float s = 0.f;
            #pragma unroll
            for (int k = 0; k < 9; ++k) s += tap[k] * wsm[oc * 9 + k];
            acc[oc] += s;
        }
    }
    #pragma unroll
    for (int i = 0; i < 27; ++i) acc[i] += b_om[i];

    // softmax over mask logits acc[18..26]
    float mx = acc[18];
    #pragma unroll
    for (int i = 19; i < 27; ++i) mx = fmaxf(mx, acc[i]);
    float ex[9], sum = 0.f;
    #pragma unroll
    for (int i = 0; i < 9; ++i) { ex[i] = __expf(acc[18 + i] - mx); sum += ex[i]; }
    const float inv = 1.f / sum;

    const int posidx = h * Ww + w;
    #pragma unroll
    for (int k = 0; k < 9; ++k) {
        size_t o = ((size_t)b * K2c + k) * (Hh * Ww) + posidx;
        dyw[o] = acc[2 * k];        // offset channel 2k   -> dy[k]
        dxw[o] = acc[2 * k + 1];    // offset channel 2k+1 -> dx[k]
        mkw[o] = ex[k] * inv;       // softmaxed mask
    }
}

// ---------------------------------------------------------------------------
// Kernel 2: convert weight (O=128, C=128, 3, 3) f32 -> f16 row-major [128][1152]
// (flattened reduction index r = c*9 + k matches the memory layout directly).
// ---------------------------------------------------------------------------
__global__ void wconv_kernel(const float* __restrict__ w, _Float16* __restrict__ wh, int n)
{
    int i = blockIdx.x * blockDim.x + threadIdx.x;
    if (i < n) wh[i] = (_Float16)w[i];
}

// ---------------------------------------------------------------------------
// Kernel 3: deformable sampling + WMMA GEMM.
// Block = 16 consecutive positions in one row (N-dim) x 128 out channels.
// 8 waves, each wave owns a 16-row slice of the output channels.
// B-matrix (1152 x 16) is built in LDS in native wave32 16-bit fragment layout.
// ---------------------------------------------------------------------------
__global__ __launch_bounds__(256) void dcn_wmma_kernel(
    const float* __restrict__ x,
    const float* __restrict__ dyw, const float* __restrict__ dxw,
    const float* __restrict__ mkw,
    const _Float16* __restrict__ Wh,
    const float* __restrict__ bias,
    float* __restrict__ out)
{
    __shared__ __align__(32) _Float16 ldsB[NKC * 32 * 16];  // 36 KB
    __shared__ int   gIdx[144][4];
    __shared__ float gWt[144][4];

    const int tid = threadIdx.x;
    const int blk = blockIdx.x;
    const int w0  = (blk & 3) << 4;   // 4 position-tiles per row
    const int h   = (blk >> 2) & 63;
    const int b0  = blk >> 8;

    // --- Phase 1: per-(tap,pos) bilinear geometry (channel independent) -----
    if (tid < 144) {
        const int k   = tid >> 4;     // tap 0..8
        const int pos = tid & 15;
        const int ky = k / 3, kx = k % 3;
        const int w = w0 + pos;
        const size_t oi = ((size_t)b0 * K2c + k) * (Hh * Ww) + h * Ww + w;
        const float dy = dyw[oi], dx = dxw[oi], mk = mkw[oi];
        const float py = (float)(h - 1 + ky) + dy;
        const float px = (float)(w - 1 + kx) + dx;
        const float fy0 = floorf(py), fx0 = floorf(px);
        const int y0 = (int)fy0, x0 = (int)fx0;
        const float wy1 = py - fy0, wx1 = px - fx0;
        const float wy0 = 1.f - wy1, wx0 = 1.f - wx1;
        const int   ys[2]  = {y0, y0 + 1};
        const int   xs[2]  = {x0, x0 + 1};
        const float wys[2] = {wy0, wy1};
        const float wxs[2] = {wx0, wx1};
        #pragma unroll
        for (int j = 0; j < 4; ++j) {
            int yy = ys[j >> 1], xx = xs[j & 1];
            bool valid = (yy >= 0) && (yy < Hh) && (xx >= 0) && (xx < Ww);
            int yc = min(max(yy, 0), Hh - 1);
            int xc = min(max(xx, 0), Ww - 1);
            gIdx[tid][j] = yc * Ww + xc;
            gWt[tid][j]  = valid ? (wys[j >> 1] * wxs[j & 1] * mk) : 0.f;
        }
    }
    __syncthreads();

    // --- Phase 2: gather + write B matrix in wave32 f16 fragment layout -----
    {
        const int c_sub = tid >> 4;   // 16 channel lanes
        const int pos   = tid & 15;
        #pragma unroll
        for (int co = 0; co < 8; ++co) {
            const int c = co * 16 + c_sub;
            const float* xc = x + ((size_t)b0 * Cch + c) * (Hh * Ww);
            #pragma unroll
            for (int k = 0; k < 9; ++k) {
                const int g = k * 16 + pos;
                float v = gWt[g][0] * xc[gIdx[g][0]] + gWt[g][1] * xc[gIdx[g][1]]
                        + gWt[g][2] * xc[gIdx[g][2]] + gWt[g][3] * xc[gIdx[g][3]];
                const int r  = c * 9 + k;          // reduction index
                const int kc = r >> 5;             // 32-wide K chunk
                const int rr = r & 31;
                // ISA 16-bit B layout: lanes 0-15 hold K{0..7,16..23} of col N,
                // lanes 16-31 hold K{8..15,24..31}.
                const int lane = pos + ((rr & 8) << 1);
                const int idx  = (rr & 7) | ((rr >> 1) & 8);
                ldsB[(kc * 32 + lane) * 16 + idx] = (_Float16)v;
            }
        }
    }
    __syncthreads();

    // --- Phase 3: 36 chained v_wmma_f32_16x16x32_f16 per wave ---------------
    const int wave = tid >> 5;
    const int lane = tid & 31;
    const int row  = wave * 16 + (lane & 15);          // output channel row
    const _Float16* Wrow = Wh + (size_t)row * RED;
    const int halfsel = (lane >> 4) << 3;              // 0 or 8 (A K-grouping)

    v8f acc = {};
    for (int kc = 0; kc < NKC; ++kc) {
        const int colbase = kc * 32 + halfsel;
        v8h alo = *(const v8h*)(Wrow + colbase);        // K {0..7} / {8..15}
        v8h ahi = *(const v8h*)(Wrow + colbase + 16);   // K {16..23} / {24..31}
        v16h a;
        #pragma unroll
        for (int i = 0; i < 8; ++i) { a[i] = alo[i]; a[i + 8] = ahi[i]; }
        v16h bfr = *(const v16h*)(ldsB + (kc * 32 + lane) * 16);
        acc = __builtin_amdgcn_wmma_f32_16x16x32_f16(
                  false, a, false, bfr, (short)0, acc, false, false);
    }

    // --- Phase 4: C/D layout -> NCHW store (+bias) --------------------------
    const int wcol  = w0 + (lane & 15);
    const int obase = wave * 16 + ((lane >> 4) << 3);  // lanes 16-31 -> M+8
    #pragma unroll
    for (int r = 0; r < 8; ++r) {
        const int o = obase + r;
        out[(((size_t)b0 * Och + o) * Hh + h) * Ww + wcol] = acc[r] + bias[o];
    }
}

// ---------------------------------------------------------------------------
extern "C" void kernel_launch(void* const* d_in, const int* in_sizes, int n_in,
                              void* d_out, int out_size, void* d_ws, size_t ws_size,
                              hipStream_t stream)
{
    (void)in_sizes; (void)n_in; (void)out_size; (void)ws_size;
    const float* x      = (const float*)d_in[0];
    const float* w_om   = (const float*)d_in[1];
    const float* b_om   = (const float*)d_in[2];
    const float* weight = (const float*)d_in[3];
    const float* bias   = (const float*)d_in[4];
    float* out = (float*)d_out;

    // workspace layout
    const int nOff = Bsz * K2c * Hh * Ww;          // 294912 floats each
    float* dyw = (float*)d_ws;
    float* dxw = dyw + nOff;
    float* mkw = dxw + nOff;
    _Float16* Wh = (_Float16*)(mkw + nOff);        // 128*1152 halfs

    // 1) offset/mask conv + softmax: 32768 positions / 256 = 128 blocks
    om_conv_kernel<<<128, 256, 0, stream>>>(x, w_om, b_om, dyw, dxw, mkw);

    // 2) weight f32 -> f16
    const int nW = Och * RED;                      // 147456
    wconv_kernel<<<(nW + 255) / 256, 256, 0, stream>>>(weight, Wh, nW);

    // 3) deformable sampling + WMMA GEMM: 8 * 64 * 4 = 2048 blocks
    dcn_wmma_kernel<<<Bsz * Hh * (Ww / 16), 256, 0, stream>>>(
        x, dyw, dxw, mkw, Wh, bias, out);
}
